// ExtractorMLP_78769700209144
// MI455X (gfx1250) — matrix operand order repaired and checked by reading, not compiled, not run on hardware
//
#include <hip/hip_runtime.h>

typedef __attribute__((ext_vector_type(16))) _Float16 v16h;
typedef __attribute__((ext_vector_type(8)))  _Float16 v8h;
typedef __attribute__((ext_vector_type(8)))  float    v8f;
typedef __attribute__((ext_vector_type(4)))  float    v4f;

#define W1T_PITCH 136   // 128 + 8 halves pad -> 272B row stride, 4-bank rotation/lane
#define W2T_PITCH 264   // 256 + 8 halves pad -> 528B row stride, 4-bank rotation/lane
#define STG_PITCH 40    // 32 + 8 halves pad  -> 80B row stride, conflict-free b128

__device__ __forceinline__ v16h lds_load16(const _Float16* p) {
  v8h lo = *(const v8h*)(p);
  v8h hi = *(const v8h*)(p + 8);
  v16h r;
#pragma unroll
  for (int i = 0; i < 8; ++i) { r[i] = lo[i]; r[i + 8] = hi[i]; }
  return r;
}

__global__ __launch_bounds__(256, 1) void extractor_mlp_kernel(
    const float* __restrict__ emb, const long long* __restrict__ eidx,
    const float* __restrict__ W1, const float* __restrict__ b1,
    const float* __restrict__ W2, const float* __restrict__ b2,
    const float* __restrict__ W3, const float* __restrict__ b3,
    float* __restrict__ out, int E, int numTiles) {
  // 69632 + 33792 + 10240 + 1024 + 256 + 256 = 115200 B LDS (of 320 KB/WGP)
  __shared__ _Float16 sW1t[256 * W1T_PITCH];   // W1^T: [n=256][k=128] f16
  __shared__ _Float16 sW2t[64 * W2T_PITCH];    // W2^T: [n=64][k=256]  f16
  __shared__ _Float16 sStg[8 * 16 * STG_PITCH];// per-wave 16x32 h1 staging
  __shared__ float sB1[256];
  __shared__ float sB2[64];
  __shared__ float sW3[64];

  const int tid = threadIdx.x;

  // ---- one-time per block: convert weights fp32 -> f16 transposed in LDS ----
  for (int idx = tid; idx < 128 * 256; idx += 256) {        // W1 [128][256] row-major
    int k = idx >> 8, n = idx & 255;                        // coalesced global read
    sW1t[n * W1T_PITCH + k] = (_Float16)W1[idx];
  }
  for (int idx = tid; idx < 256 * 64; idx += 256) {         // W2 [256][64]
    int k = idx >> 6, n = idx & 63;
    sW2t[n * W2T_PITCH + k] = (_Float16)W2[idx];
  }
  sB1[tid] = b1[tid];
  if (tid < 64) { sB2[tid] = b2[tid]; sW3[tid] = W3[tid]; }
  __syncthreads();

  const int wave = tid >> 5;
  const int lane = tid & 31;
  const int nl   = lane & 15;   // column / row-within-tile id
  const int s    = lane >> 4;   // half-wave side
  const float b3v = b3[0];
  _Float16* stg = &sStg[wave * 16 * STG_PITCH];

  const v8f zero = {};

  for (int tile = blockIdx.x * 8 + wave; tile < numTiles; tile += gridDim.x * 8) {
    // ---- gather 16 edges' concat features as WMMA A-fragments (f16) ----
    int e  = (tile << 4) + nl;
    int eg = e < E ? e : (E - 1);
    const int ci = (int)eidx[eg];         // col endpoint
    const int ri = (int)eidx[E + eg];     // row endpoint
    const float* f0 = emb + (long long)ci * 64;
    const float* f1 = emb + (long long)ri * 64;

    v16h a[4];
#pragma unroll
    for (int c = 0; c < 4; ++c) {         // K chunks of 32 (total K=128)
      const float* p  = (c < 2) ? f0 : f1;
      const int   off = ((c & 1) << 5) + (s << 4);
      v4f x0 = *(const v4f*)(p + off);
      v4f x1 = *(const v4f*)(p + off + 4);
      v4f x2 = *(const v4f*)(p + off + 8);
      v4f x3 = *(const v4f*)(p + off + 12);
      v16h av;
#pragma unroll
      for (int j = 0; j < 4; ++j) {
        av[j]      = (_Float16)x0[j];
        av[j + 4]  = (_Float16)x1[j];
        av[j + 8]  = (_Float16)x2[j];
        av[j + 12] = (_Float16)x3[j];
      }
      a[c] = av;
    }

    v8f acc2[4];
#pragma unroll
    for (int t2 = 0; t2 < 4; ++t2) acc2[t2] = zero;
    float acc3[8] = {0.f, 0.f, 0.f, 0.f, 0.f, 0.f, 0.f, 0.f};

    // ---- layer1 interleaved with layer2: q = layer-2 K-chunk (32 wide) ----
#pragma unroll
    for (int q = 0; q < 8; ++q) {
#pragma unroll
      for (int h = 0; h < 2; ++h) {       // two layer-1 N-tiles per chunk
        const int t = q * 2 + h;
        v8f acc = zero;
#pragma unroll
        for (int c = 0; c < 4; ++c) {
          v16h bv = lds_load16(&sW1t[(t * 16 + nl) * W1T_PITCH + c * 32 + s * 16]);
          acc = __builtin_amdgcn_wmma_f32_16x16x32_f16(false, a[c], false, bv,
                                                       (short)0, acc, false, false);
        }
        const float b1v = sB1[t * 16 + nl];
#pragma unroll
        for (int r = 0; r < 8; ++r) {     // relu(bias+acc) -> f16 staging
          float v = acc[r] + b1v;
          v = v > 0.0f ? v : 0.0f;
          stg[(r + s * 8) * STG_PITCH + h * 16 + nl] = (_Float16)v;
        }
      }
      // reload staged h1 chunk as layer-2 A-fragment (same-wave DS: in order)
      v16h a2;
      {
        v8h lo = *(const v8h*)&stg[nl * STG_PITCH + s * 16];
        v8h hi = *(const v8h*)&stg[nl * STG_PITCH + s * 16 + 8];
#pragma unroll
        for (int i = 0; i < 8; ++i) { a2[i] = lo[i]; a2[i + 8] = hi[i]; }
      }
#pragma unroll
      for (int t2 = 0; t2 < 4; ++t2) {
        v16h bv = lds_load16(&sW2t[(t2 * 16 + nl) * W2T_PITCH + q * 32 + s * 16]);
        acc2[t2] = __builtin_amdgcn_wmma_f32_16x16x32_f16(false, a2, false, bv,
                                                          (short)0, acc2[t2], false, false);
      }
    }

    // ---- layer3: relu(h2+b2) . W3, reduce across the 16 columns ----
#pragma unroll
    for (int t2 = 0; t2 < 4; ++t2) {
      const float b2v = sB2[t2 * 16 + nl];
      const float w3v = sW3[t2 * 16 + nl];
#pragma unroll
      for (int r = 0; r < 8; ++r) {
        float v = acc2[t2][r] + b2v;
        v = v > 0.0f ? v : 0.0f;
        acc3[r] += v * w3v;
      }
    }
#pragma unroll
    for (int off = 1; off < 16; off <<= 1) {
#pragma unroll
      for (int r = 0; r < 8; ++r) acc3[r] += __shfl_xor(acc3[r], off, 32);
    }
    if (nl == 0) {                         // lane 0 -> rows 0..7, lane 16 -> 8..15
      const int base = (tile << 4) + (s << 3);
#pragma unroll
      for (int r = 0; r < 8; ++r)
        if (base + r < E) out[base + r] = acc3[r] + b3v;
    }
  }
}

extern "C" void kernel_launch(void* const* d_in, const int* in_sizes, int n_in,
                              void* d_out, int out_size, void* d_ws, size_t ws_size,
                              hipStream_t stream) {
  (void)in_sizes; (void)n_in; (void)d_ws; (void)ws_size;
  const float*     emb  = (const float*)d_in[0];
  const long long* eidx = (const long long*)d_in[1];
  // d_in[2] = batch: unused by the reference computation
  const float* W1 = (const float*)d_in[3];
  const float* b1 = (const float*)d_in[4];
  const float* W2 = (const float*)d_in[5];
  const float* b2 = (const float*)d_in[6];
  const float* W3 = (const float*)d_in[7];
  const float* b3 = (const float*)d_in[8];
  float* out = (float*)d_out;

  const int E = out_size;                  // [E,1] fp32 logits
  const int tiles = (E + 15) >> 4;
  int blocks = (tiles + 7) / 8;
  if (blocks > 1250) blocks = 1250;        // 8 waves/block, grid-stride over tiles

  hipLaunchKernelGGL(extractor_mlp_kernel, dim3(blocks), dim3(256), 0, stream,
                     emb, eidx, W1, b1, W2, b2, W3, b3, out, E, tiles);
}
